// HybridRumourModel_32993938768203
// MI455X (gfx1250) — matrix operand order repaired and checked by reading, not compiled, not run on hardware
//
#include <hip/hip_runtime.h>
#include <hip/hip_bf16.h>
#include <math.h>

// Problem constants (match reference)
#define N_NODES   100000
#define N_EDGES   3200000
#define F_INF     256
#define HDIM      64
#define G_GRAPHS  1000
#define C_CLASSES 2
#define NT        4     // N-tiles (16 cols each) per wave in the WMMA GEMM

typedef __attribute__((ext_vector_type(16))) _Float16 v16h;
typedef __attribute__((ext_vector_type(8)))  float    v8f;

// ---------------------------------------------------------------------------
// B-fragment swizzle: weights stored as f16 in this GEMM's per-lane fragment layout (contiguous k).
// half index = ((kt*ntiles + nt)*32 + lane)*16 + j
//   maps to logical B[k][n] with k = kt*32 + (lane>=16 ? 16 : 0) + j,
//                               n = nt*16 + (lane & 15)
// so the GEMM loads each lane's 16 B values as one contiguous 32B chunk.
// ---------------------------------------------------------------------------
__device__ __forceinline__ void frag_decode(int o, int N, int& k, int& n) {
  int j    = o & 15;
  int lane = (o >> 4) & 31;
  int tile = o >> 9;
  int ntiles = N >> 4;
  int nt = tile % ntiles;
  int kt = tile / ntiles;
  k = kt * 32 + ((lane >> 4) << 4) + j;
  n = nt * 16 + (lane & 15);
}

// Segments (half counts):
//  [0,16384)      lin_in_w [K=256,N=64]
//  [16384,28672)  ggnn_w   3 x [64,64]
//  [28672,40960)  w_ih^T   [64,192]   (src is [192,64])
//  [40960,53248)  w_hh^T   [64,192]
//  [53248,57344)  gat_w    [64,64]
//  [57344,61440)  lin1_w   [64,64]
__global__ void prep_weights(const float* __restrict__ lin_in_w,
                             const float* __restrict__ ggnn_w,
                             const float* __restrict__ w_ih,
                             const float* __restrict__ w_hh,
                             const float* __restrict__ gat_w,
                             const float* __restrict__ lin1_w,
                             _Float16* __restrict__ wb) {
  int i = blockIdx.x * blockDim.x + threadIdx.x;
  if (i >= 61440) return;
  float v; int k, n;
  if (i < 16384)      { frag_decode(i, 64, k, n);              v = lin_in_w[k * 64 + n]; }
  else if (i < 28672) { int l = i - 16384; int g = l >> 12;
                        frag_decode(l & 4095, 64, k, n);       v = ggnn_w[g * 4096 + k * 64 + n]; }
  else if (i < 40960) { frag_decode(i - 28672, 192, k, n);     v = w_ih[n * 64 + k]; }
  else if (i < 53248) { frag_decode(i - 40960, 192, k, n);     v = w_hh[n * 64 + k]; }
  else if (i < 57344) { frag_decode(i - 53248, 64, k, n);      v = gat_w[k * 64 + n]; }
  else                { frag_decode(i - 57344, 64, k, n);      v = lin1_w[k * 64 + n]; }
  wb[i] = (_Float16)v;
}

// ---------------------------------------------------------------------------
// WMMA GEMM: C[M,N] = act(A[M,K] * Bsw + bias[N])
// A fp32 row-major (float4 loads, converted to f16 in-register).
// Bsw: f16 in fragment layout (see above). One wave computes a 16x(NT*16)
// strip of C: A fragment loaded once per K-step, 4 B fragments loaded with
// constant immediate offsets (1024B apart), then 4 back-to-back WMMAs.
// Requires N % 64 == 0, K % 32 == 0.
// ---------------------------------------------------------------------------
__global__ void wmma_gemm(const float* __restrict__ A, const _Float16* __restrict__ Bsw,
                          const float* __restrict__ bias, float* __restrict__ C,
                          int M, int N, int K, int act) {
  int wave = (blockIdx.x * blockDim.x + threadIdx.x) >> 5;
  int lane = threadIdx.x & 31;
  int ntiles  = N >> 4;
  int ngroups = ntiles >> 2;             // groups of NT=4 tiles
  int mt = wave / ngroups;
  int ng = wave - mt * ngroups;
  int row0 = mt << 4;
  if (row0 >= M) return;                 // wave-uniform: EXEC all-1s at WMMA
  int half = lane >> 4;
  int l16  = lane & 15;
  int arow = row0 + l16;
  if (arow >= M) arow = M - 1;           // dup row for tail; never stored

  v8f c[NT] = {};
  int ktiles = K >> 5;
  const float* arow_base = A + (size_t)arow * K + half * 16;
  const v16h* bbase = (const v16h*)Bsw + (size_t)ng * NT * 32 + lane;
  for (int kt = 0; kt < ktiles; ++kt) {
    // A fragment: row l16,
    //   elems 0..15 -> K = kt*32 + half*16 + (0..15)
    const float* arow_p = arow_base + kt * 32;
    if (kt + 1 < ktiles) __builtin_prefetch(arow_p + 32, 0, 1);  // global_prefetch_b8
    const float4* ap = (const float4*)arow_p;
    float4 a0 = ap[0], a1 = ap[1], a2 = ap[2], a3 = ap[3];
    v16h a;
    a[0]  = (_Float16)a0.x; a[1]  = (_Float16)a0.y; a[2]  = (_Float16)a0.z; a[3]  = (_Float16)a0.w;
    a[4]  = (_Float16)a1.x; a[5]  = (_Float16)a1.y; a[6]  = (_Float16)a1.z; a[7]  = (_Float16)a1.w;
    a[8]  = (_Float16)a2.x; a[9]  = (_Float16)a2.y; a[10] = (_Float16)a2.z; a[11] = (_Float16)a2.w;
    a[12] = (_Float16)a3.x; a[13] = (_Float16)a3.y; a[14] = (_Float16)a3.z; a[15] = (_Float16)a3.w;
    // 4 B fragments: single base, 1024B constant strides -> immediate offsets
    const v16h* bp = bbase + (size_t)kt * ntiles * 32;
    v16h b0 = bp[0];
    v16h b1 = bp[32];
    v16h b2 = bp[64];
    v16h b3 = bp[96];
    c[0] = __builtin_amdgcn_wmma_f32_16x16x32_f16(false, a, false, b0, (short)0, c[0], false, false);
    c[1] = __builtin_amdgcn_wmma_f32_16x16x32_f16(false, a, false, b1, (short)0, c[1], false, false);
    c[2] = __builtin_amdgcn_wmma_f32_16x16x32_f16(false, a, false, b2, (short)0, c[2], false, false);
    c[3] = __builtin_amdgcn_wmma_f32_16x16x32_f16(false, a, false, b3, (short)0, c[3], false, false);
  }

  // D layout: lane l, reg r -> row = row0 + r + half*8, col = tile*16 + l16
  bool full = (row0 + 16 <= M);
#pragma unroll
  for (int t = 0; t < NT; ++t) {
    int colt = (ng * NT + t) * 16 + l16;
    float bv = bias ? bias[colt] : 0.f;
    float* cp = C + (size_t)(row0 + half * 8) * N + colt;
    if (full) {
#pragma unroll
      for (int r = 0; r < 8; ++r) {
        float v = c[t][r] + bv;
        if (act == 1) v = fmaxf(v, 0.f);
        cp[(size_t)r * N] = v;
      }
    } else {
      for (int r = 0; r < 8; ++r) {
        int row = row0 + half * 8 + r;
        if (row < M) {
          float v = c[t][r] + bv;
          if (act == 1) v = fmaxf(v, 0.f);
          C[(size_t)row * N + colt] = v;
        }
      }
    }
  }
}

// ---------------------------------------------------------------------------
// Elementwise / scatter kernels
// ---------------------------------------------------------------------------
__global__ void fill_f32(float* __restrict__ p, int n, float v) {
  int i = blockIdx.x * blockDim.x + threadIdx.x;
  if (i < n) p[i] = v;
}

__global__ void relu_inplace(float* __restrict__ p, int n) {
  int i = blockIdx.x * blockDim.x + threadIdx.x;
  if (i < n) { float v = p[i]; p[i] = v > 0.f ? v : 0.f; }
}

// m[dst] += xw[src] : one thread per (edge, feature)
__global__ void edge_scatter(const int* __restrict__ src, const int* __restrict__ dst,
                             const float* __restrict__ xw, float* __restrict__ m) {
  int i = blockIdx.x * blockDim.x + threadIdx.x;
  if (i >= N_EDGES * HDIM) return;
  int e = i >> 6, f = i & 63;
  atomicAdd(&m[(size_t)dst[e] * HDIM + f], xw[(size_t)src[e] * HDIM + f]);
}

// GRU cell elementwise (gi, gh are [N,192]; h updated in place)
__global__ void gru_elem(const float* __restrict__ gi, const float* __restrict__ gh,
                         float* __restrict__ h) {
  int i = blockIdx.x * blockDim.x + threadIdx.x;
  if (i >= N_NODES * HDIM) return;
  int n = i >> 6, f = i & 63;
  const float* gin = gi + (size_t)n * 192;
  const float* ghn = gh + (size_t)n * 192;
  float r  = 1.f / (1.f + expf(-(gin[f] + ghn[f])));
  float z  = 1.f / (1.f + expf(-(gin[64 + f] + ghn[64 + f])));
  float nn = tanhf(gin[128 + f] + r * ghn[128 + f]);
  h[i] = (1.f - z) * nn + z * h[i];
}

// Per-node attention scores a_s = xw . att_src, a_d = xw . att_dst
__global__ void gat_scores(const float* __restrict__ xw, const float* __restrict__ att_src,
                           const float* __restrict__ att_dst,
                           float* __restrict__ a_s, float* __restrict__ a_d) {
  int n = blockIdx.x * blockDim.x + threadIdx.x;
  if (n >= N_NODES) return;
  const float* row = xw + (size_t)n * HDIM;
  float s = 0.f, d = 0.f;
#pragma unroll 8
  for (int k = 0; k < HDIM; ++k) { float v = row[k]; s += v * att_src[k]; d += v * att_dst[k]; }
  a_s[n] = s; a_d[n] = d;
}

// Pass 1: e_exp = exp(leaky_relu(a_s[s]+a_d[d])); denom[d] += e_exp
// (softmax is shift-invariant; scores are tiny so no max-subtraction needed)
__global__ void gat_pass1(const int* __restrict__ src, const int* __restrict__ dst,
                          const float* __restrict__ a_s, const float* __restrict__ a_d,
                          float* __restrict__ e_exp, float* __restrict__ denom) {
  int i = blockIdx.x * blockDim.x + threadIdx.x;
  if (i >= N_EDGES + N_NODES) return;
  int s, d;
  if (i < N_EDGES) { s = src[i]; d = dst[i]; } else { s = d = i - N_EDGES; }
  float e = a_s[s] + a_d[d];
  e = e > 0.f ? e : 0.2f * e;
  float w = expf(e);
  e_exp[i] = w;
  atomicAdd(&denom[d], w);
}

// Pass 2: out[d] += (e_exp / denom[d]) * xw[s]
__global__ void gat_pass2(const int* __restrict__ src, const int* __restrict__ dst,
                          const float* __restrict__ xw, const float* __restrict__ e_exp,
                          const float* __restrict__ denom, float* __restrict__ out) {
  int i = blockIdx.x * blockDim.x + threadIdx.x;
  if (i >= (N_EDGES + N_NODES) * HDIM) return;
  int ed = i >> 6, f = i & 63;
  int s, d;
  if (ed < N_EDGES) { s = src[ed]; d = dst[ed]; } else { s = d = ed - N_EDGES; }
  float alpha = e_exp[ed] / (denom[d] + 1e-16f);
  atomicAdd(&out[(size_t)d * HDIM + f], alpha * xw[(size_t)s * HDIM + f]);
}

__global__ void gat_finish(const float* __restrict__ out, const float* __restrict__ gat_b,
                           float* __restrict__ h) {
  int i = blockIdx.x * blockDim.x + threadIdx.x;
  if (i >= N_NODES * HDIM) return;
  float v = out[i] + gat_b[i & 63];
  h[i] = v > 0.f ? v : 0.f;
}

__global__ void pool_scatter(const float* __restrict__ h, const int* __restrict__ batch,
                             float* __restrict__ pooled, float* __restrict__ counts) {
  int i = blockIdx.x * blockDim.x + threadIdx.x;
  if (i >= N_NODES * HDIM) return;
  int n = i >> 6, f = i & 63;
  int g = batch[n];
  atomicAdd(&pooled[(size_t)g * HDIM + f], h[i]);
  if (f == 0) atomicAdd(&counts[g], 1.0f);
}

__global__ void pool_div(float* __restrict__ pooled, const float* __restrict__ counts) {
  int i = blockIdx.x * blockDim.x + threadIdx.x;
  if (i >= G_GRAPHS * HDIM) return;
  float c = counts[i >> 6];
  if (c < 1.f) c = 1.f;
  pooled[i] /= c;
}

// Final tiny layer + log_softmax over 2 classes
__global__ void head(const float* __restrict__ hidden, const float* __restrict__ w2,
                     const float* __restrict__ b2, float* __restrict__ out) {
  int g = blockIdx.x * blockDim.x + threadIdx.x;
  if (g >= G_GRAPHS) return;
  const float* hr = hidden + (size_t)g * HDIM;
  float l0 = b2[0], l1 = b2[1];
#pragma unroll 8
  for (int k = 0; k < HDIM; ++k) { float v = hr[k]; l0 += v * w2[k * 2]; l1 += v * w2[k * 2 + 1]; }
  float m = l0 > l1 ? l0 : l1;
  float lse = m + logf(expf(l0 - m) + expf(l1 - m));
  out[g * 2]     = l0 - lse;
  out[g * 2 + 1] = l1 - lse;
}

// ---------------------------------------------------------------------------
// Host orchestration
// ---------------------------------------------------------------------------
static inline void launch_gemm(const float* A, const _Float16* Bsw, const float* bias,
                               float* C, int M, int N, int K, int act, hipStream_t s) {
  int tilesM  = (M + 15) / 16;
  int ngroups = (N / 16) / NT;
  long waves  = (long)tilesM * ngroups;
  int blocks  = (int)((waves + 3) / 4);       // 4 waves (128 threads) per block
  wmma_gemm<<<blocks, 128, 0, s>>>(A, Bsw, bias, C, M, N, K, act);
}

static inline int nb(long n) { return (int)((n + 255) / 256); }

extern "C" void kernel_launch(void* const* d_in, const int* in_sizes, int n_in,
                              void* d_out, int out_size, void* d_ws, size_t ws_size,
                              hipStream_t stream) {
  const float* x        = (const float*)d_in[0];
  const int*   eidx     = (const int*)  d_in[1];
  const int*   batch    = (const int*)  d_in[2];
  const float* lin_in_w = (const float*)d_in[3];
  const float* lin_in_b = (const float*)d_in[4];
  const float* ggnn_w   = (const float*)d_in[5];
  const float* w_ih     = (const float*)d_in[6];
  const float* w_hh     = (const float*)d_in[7];
  const float* b_ih     = (const float*)d_in[8];
  const float* b_hh     = (const float*)d_in[9];
  const float* gat_w    = (const float*)d_in[10];
  const float* att_src  = (const float*)d_in[11];
  const float* att_dst  = (const float*)d_in[12];
  const float* gat_b    = (const float*)d_in[13];
  const float* lin1_w   = (const float*)d_in[14];
  const float* lin1_b   = (const float*)d_in[15];
  const float* lin2_w   = (const float*)d_in[16];
  const float* lin2_b   = (const float*)d_in[17];
  const int* src = eidx;
  const int* dst = eidx + N_EDGES;

  // Workspace arena (floats): h | buf1 | buf2 | gi | gh | f16 weights
  float* ws = (float*)d_ws;
  const size_t N64  = (size_t)N_NODES * HDIM;
  const size_t N192 = (size_t)N_NODES * 192;
  float* h    = ws;
  float* buf1 = h + N64;              // xw (GGNN msgs / GAT xw)
  float* buf2 = buf1 + N64;           // m (GGNN) / out (GAT)
  float* gi   = buf2 + N64;           // [N,192] GRU input gates; reused for GAT temps
  float* gh   = gi + N192;            // [N,192] GRU hidden gates; reused for pool temps
  _Float16* wb = (_Float16*)(gh + N192);

  // GAT temps alias gi region (free during GAT)
  float* a_s   = gi;
  float* a_d   = gi + N_NODES;
  float* denom = gi + 2 * (size_t)N_NODES;
  float* e_exp = gi + 3 * (size_t)N_NODES;     // E+N = 3.3M <= N*192 - 3N
  // Pool temps alias gh region (free after GRU)
  float* pooled = gh;
  float* counts = gh + (size_t)G_GRAPHS * HDIM;
  float* hidden = counts + G_GRAPHS;

  _Float16* wb_lin_in = wb;
  _Float16* wb_ggnn   = wb + 16384;
  _Float16* wb_ihT    = wb + 28672;
  _Float16* wb_hhT    = wb + 40960;
  _Float16* wb_gat    = wb + 53248;
  _Float16* wb_lin1   = wb + 57344;

  prep_weights<<<nb(61440), 256, 0, stream>>>(lin_in_w, ggnn_w, w_ih, w_hh, gat_w, lin1_w, wb);

  // h = relu(x @ lin_in_w + lin_in_b)
  launch_gemm(x, wb_lin_in, lin_in_b, h, N_NODES, HDIM, F_INF, 1, stream);

  // 3x GatedGraphConv steps
  for (int step = 0; step < 3; ++step) {
    launch_gemm(h, wb_ggnn + step * 4096, nullptr, buf1, N_NODES, HDIM, HDIM, 0, stream);
    fill_f32<<<nb(N64), 256, 0, stream>>>(buf2, (int)N64, 0.f);
    edge_scatter<<<nb((long)N_EDGES * HDIM), 256, 0, stream>>>(src, dst, buf1, buf2);
    launch_gemm(buf2, wb_ihT, b_ih, gi, N_NODES, 192, HDIM, 0, stream);
    launch_gemm(h,    wb_hhT, b_hh, gh, N_NODES, 192, HDIM, 0, stream);
    gru_elem<<<nb(N64), 256, 0, stream>>>(gi, gh, h);
  }
  relu_inplace<<<nb(N64), 256, 0, stream>>>(h, (int)N64);

  // GATConv
  launch_gemm(h, wb_gat, nullptr, buf1, N_NODES, HDIM, HDIM, 0, stream);
  gat_scores<<<nb(N_NODES), 256, 0, stream>>>(buf1, att_src, att_dst, a_s, a_d);
  fill_f32<<<nb(N_NODES), 256, 0, stream>>>(denom, N_NODES, 0.f);
  fill_f32<<<nb(N64), 256, 0, stream>>>(buf2, (int)N64, 0.f);
  gat_pass1<<<nb((long)(N_EDGES + N_NODES)), 256, 0, stream>>>(src, dst, a_s, a_d, e_exp, denom);
  gat_pass2<<<nb((long)(N_EDGES + N_NODES) * HDIM), 256, 0, stream>>>(src, dst, buf1, e_exp, denom, buf2);
  gat_finish<<<nb(N64), 256, 0, stream>>>(buf2, gat_b, h);

  // global_mean_pool
  fill_f32<<<nb(G_GRAPHS * HDIM + G_GRAPHS), 256, 0, stream>>>(pooled, G_GRAPHS * HDIM + G_GRAPHS, 0.f);
  pool_scatter<<<nb(N64), 256, 0, stream>>>(h, batch, pooled, counts);
  pool_div<<<nb(G_GRAPHS * HDIM), 256, 0, stream>>>(pooled, counts);

  // classifier head
  launch_gemm(pooled, wb_lin1, lin1_b, hidden, G_GRAPHS, HDIM, HDIM, 1, stream);
  head<<<nb(G_GRAPHS), 256, 0, stream>>>(hidden, lin2_w, lin2_b, (float*)d_out);
}
